// _FFTConvForward_81681688035753
// MI455X (gfx1250) — compile-verified, matrix operand order checked by reading, not run on hardware
//
#include <hip/hip_runtime.h>

// ---- CDNA5 WMMA vector types -------------------------------------------------
typedef __bf16 v2bf  __attribute__((ext_vector_type(2)));
typedef __bf16 v8bf  __attribute__((ext_vector_type(8)));
typedef __bf16 v16bf __attribute__((ext_vector_type(16)));
typedef float  v8f   __attribute__((ext_vector_type(8)));

// ---- problem constants -------------------------------------------------------
#define C_IN   64
#define C_OUT  64
#define HDIM   256
#define WDIM   256
#define BATCH  16
#define KH     7
#define KW     7
#define NTAPS  49
#define PAD    3
#define XT     32               // output x per block
#define YT     8                // output rows per block
#define PC     (XT + KW - 1)    // 38 patch cols
#define PR     (YT + KH - 1)    // 14 patch rows
#define CSTR   72               // padded channel stride: 144B lane stride -> conflict-free
#define GTAPS  4                // taps staged per group
#define SW_BYTES   (GTAPS * C_OUT * CSTR * 2)        // 36864
#define SIN_ELEMS  (PR * PC * CSTR)                  // 38304
#define LDS_BYTES  (SW_BYTES + SIN_ELEMS * 2)        // 113472

// ---- gfx1250 async global->LDS copy (bypasses VGPRs, tracked by ASYNCcnt) ---
__device__ __forceinline__ void async_b128_to_lds(unsigned lds_off, const void* g) {
  asm volatile("global_load_async_to_lds_b128 %0, %1, off"
               :: "v"(lds_off), "v"((unsigned long long)(size_t)g) : "memory");
}
__device__ __forceinline__ void wait_async0() {
  asm volatile("s_wait_asynccnt 0x0" ::: "memory");
}

// load a 16-bit WMMA fragment: lane-half picks K-offset {0,8}, halves at +0/+16
__device__ __forceinline__ v16bf frag16(const __bf16* p) {
  const v8bf* q = (const v8bf*)p;
  v8bf lo = q[0], hi = q[2];
  v16bf r;
#pragma unroll
  for (int t = 0; t < 8; ++t) { r[t] = lo[t]; r[t + 8] = hi[t]; }
  return r;
}

// ---- prep: w[o][c][i][j] f32  ->  wsT[tap][o*64+c] bf16 (L2-resident) -------
__global__ __launch_bounds__(256)
void prep_weights(const float* __restrict__ wgt, __bf16* __restrict__ wsT) {
  const int idx = blockIdx.x * 256 + threadIdx.x;   // = tap*4096 + oc
  const int tap = idx >> 12;
  const int oc  = idx & 4095;
  wsT[idx] = (__bf16)wgt[(size_t)oc * NTAPS + tap];
}

// ---- main: implicit-GEMM 7x7 conv, bf16 WMMA, f32 accumulate ----------------
__global__ __launch_bounds__(256)
void conv7x7_wmma(const float* __restrict__ sig, const __bf16* __restrict__ wsT,
                  const float* __restrict__ bias, float* __restrict__ out) {
  extern __shared__ __align__(16) unsigned char smem[];
  __bf16* sW  = (__bf16*)smem;                   // [GTAPS][C_OUT][CSTR]
  __bf16* sIn = (__bf16*)(smem + SW_BYTES);      // [PR][PC][CSTR]

  const int tid = threadIdx.x;
  const int x0  = blockIdx.x * XT;
  const int y0  = blockIdx.y * YT;
  const int b   = blockIdx.z;

  // ---- stage input patch: 2 channels packed per ds_store_b32 ----------------
  for (int f = tid; f < (C_IN / 2) * PR * PC; f += 256) {
    int cp  = f / (PR * PC);
    int pos = f - cp * (PR * PC);
    int r   = pos / PC;
    int col = pos - r * PC;
    int rg  = y0 + r - PAD;
    int cg  = x0 + col - PAD;
    float v0 = 0.f, v1 = 0.f;
    if ((unsigned)rg < (unsigned)HDIM && (unsigned)cg < (unsigned)WDIM) {
      size_t base = (((size_t)b * C_IN + 2 * cp) * HDIM + rg) * WDIM + cg;
      v0 = sig[base];
      v1 = sig[base + (size_t)HDIM * WDIM];
    }
    v2bf pk = {(__bf16)v0, (__bf16)v1};
    *(v2bf*)&sIn[pos * CSTR + 2 * cp] = pk;
  }

  // ---- wave / lane mapping: 2 M-tiles x 4 N-tiles per wave ------------------
  const int lane = tid & 31;
  const int l    = lane & 15;
  const int half = lane >> 4;
  const int wave = tid >> 5;
  const int mp   = (wave & 1) * 2;       // M-tile pair base (tiles mp, mp+1)
  const int rp   = (wave >> 1) * 2;      // output row pair base (rows rp, rp+1)

  v8f acc[2][2][2];                      // [mi][ri][xt]
#pragma unroll
  for (int mi = 0; mi < 2; ++mi)
#pragma unroll
    for (int ri = 0; ri < 2; ++ri)
#pragma unroll
      for (int xt = 0; xt < 2; ++xt)
        acc[mi][ri][xt] = v8f{0.f, 0.f, 0.f, 0.f, 0.f, 0.f, 0.f, 0.f};

  // async-stage ntap weight planes (contiguous bf16 stream -> padded LDS rows)
  auto stageW = [&](int g, int ntap) {
    const int nchunk = ntap * C_OUT * 8;         // 16B chunks
    for (int ch = tid; ch < nchunk; ch += 256) {
      int t   = ch >> 9;
      int rem = ch & 511;
      int o   = rem >> 3;
      int k   = rem & 7;
      unsigned lds = (unsigned)(size_t)&sW[(t * C_OUT + o) * CSTR + k * 8];
      const void* gp = &wsT[(size_t)(GTAPS * g + t) * 4096 + o * 64 + k * 8];
      async_b128_to_lds(lds, gp);
    }
  };

  auto computeTap = [&](int t, int tap) {
    const int i = tap / KW, j = tap - i * KW;
#pragma unroll
    for (int cb = 0; cb < C_IN; cb += 32) {
      const int ko = cb + 8 * half;
      // -- preload all fragments into distinct regs: lets the scheduler issue
      //    12 ds_load_b128 back-to-back and use staggered s_wait_dscnt.
      v16bf a[2], bm[2][2];
#pragma unroll
      for (int mi = 0; mi < 2; ++mi)
        a[mi] = frag16(&sW[(t * C_OUT + (mp + mi) * 16 + l) * CSTR + ko]);
#pragma unroll
      for (int ri = 0; ri < 2; ++ri)
#pragma unroll
        for (int xt = 0; xt < 2; ++xt)
          bm[ri][xt] = frag16(&sIn[((rp + ri + i) * PC + xt * 16 + l + j) * CSTR + ko]);
      // -- 8 WMMAs; each B reused twice from registers, each A four times.
#pragma unroll
      for (int ri = 0; ri < 2; ++ri)
#pragma unroll
        for (int xt = 0; xt < 2; ++xt)
#pragma unroll
          for (int mi = 0; mi < 2; ++mi)
            acc[mi][ri][xt] = __builtin_amdgcn_wmma_f32_16x16x32_bf16(
                false, a[mi], false, bm[ri][xt], (short)0, acc[mi][ri][xt],
                false, false);
    }
  };

  // ---- 12 groups of 4 taps + 1 tail tap -------------------------------------
  for (int g = 0; g < 12; ++g) {
    stageW(g, GTAPS);
    wait_async0();
    __syncthreads();               // weights + (first iter) input visible
#pragma unroll
    for (int t = 0; t < GTAPS; ++t) computeTap(t, GTAPS * g + t);
    __syncthreads();               // protect sW reuse (WAR)
  }
  stageW(12, 1);
  wait_async0();
  __syncthreads();
  computeTap(0, 48);

  // ---- epilogue: C layout lane->N, vgpr->M (halves split M by 8) ------------
#pragma unroll
  for (int mi = 0; mi < 2; ++mi)
#pragma unroll
    for (int ri = 0; ri < 2; ++ri) {
      const int yo = y0 + rp + ri;
#pragma unroll
      for (int xt = 0; xt < 2; ++xt) {
        const int xo = x0 + xt * 16 + l;
#pragma unroll
        for (int p = 0; p < 8; ++p) {
          const int o = (mp + mi) * 16 + 8 * half + p;
          out[(((size_t)b * C_OUT + o) * HDIM + yo) * WDIM + xo] =
              acc[mi][ri][xt][p] + bias[o];
        }
      }
    }
}

extern "C" void kernel_launch(void* const* d_in, const int* in_sizes, int n_in,
                              void* d_out, int out_size, void* d_ws, size_t ws_size,
                              hipStream_t stream) {
  const float* sig  = (const float*)d_in[0];  // [16,64,256,256] f32
  const float* wgt  = (const float*)d_in[1];  // [64,64,7,7]    f32
  const float* bias = (const float*)d_in[2];  // [64]           f32
  float* out  = (float*)d_out;                // [16,64,256,256] f32
  __bf16* wsT = (__bf16*)d_ws;                // 49*4096 bf16 = 392 KB scratch

  hipLaunchKernelGGL(prep_weights, dim3((NTAPS * 4096) / 256), dim3(256), 0, stream,
                     wgt, wsT);
  hipLaunchKernelGGL(conv7x7_wmma, dim3(WDIM / XT, HDIM / YT, BATCH), dim3(256),
                     LDS_BYTES, stream, sig, wsT, bias, out);
}
